// AdaptiveEMA_65463891525758
// MI455X (gfx1250) — matrix operand order repaired and checked by reading, not compiled, not run on hardware
//
#include <hip/hip_runtime.h>

typedef __attribute__((ext_vector_type(16))) _Float16 v16h;
typedef __attribute__((ext_vector_type(4)))  _Float16 v4h;
typedef __attribute__((ext_vector_type(8)))  float    v8f;
typedef __attribute__((ext_vector_type(4)))  float    v4f;

#define B_      32
#define F_      256
#define S_      4096
#define HALO    200              // K-1 taps of history
#define CHUNK   512              // output timesteps per workgroup
#define NCHUNK  (S_ / CHUNK)     // 8
#define WCOLS   (CHUNK + HALO)   // 712 valid window columns (multiple of 4)
#define PITCH   728              // LDS row pitch (halves); 16B multiple; dword stride 364 (44 mod 64 -> 16 lanes on distinct banks)
#define KTILES  7                // 7*32 = 224 >= 216 (padded K)
#define NWAVES  8
#define BLOCK   256
#define NTILES  ((CHUNK / 16) * 2)        // 64 tiles per block
#define TILES_PER_WAVE (NTILES / NWAVES)  // 8 (compile-time uniform trip count)
#define APITCH  120              // A-frag table per-lane stride (halves); 60-dword stride -> conflict-free

#if __has_builtin(__builtin_amdgcn_exp2f)
#define FAST_EXP2(x) __builtin_amdgcn_exp2f(x)
#else
#define FAST_EXP2(x) exp2f(x)
#endif

__global__ __launch_bounds__(BLOCK)
void AdaptiveEMA_wmma_kernel(const float* __restrict__ x,
                             const float* __restrict__ log_halflife,
                             float* __restrict__ out)
{
    __shared__ _Float16 xs[B_ * PITCH];        // 46592 B: f16 input window
    __shared__ float    rden[CHUNK];           //  2048 B: reciprocal running weights per timestep
    __shared__ _Float16 afrag_lds[32 * APITCH];//  7680 B: A fragments in register layout [lane][kt][16]

    const int f      = blockIdx.x / NCHUNK;
    const int chunk  = blockIdx.x % NCHUNK;
    const int t0base = chunk * CHUNK;
    const int tid    = threadIdx.x;
    const int lane   = tid & 31;
    const int wave   = tid >> 5;

    // alpha = 0.5^(1/halflife)  =>  log2(alpha) = -1/halflife = -exp(-log_halflife)
    const float log2a = -expf(-log_halflife[f]);
    const float ln_a  = log2a * 0.69314718055994530942f;
    const float em1a  = expm1f(ln_a);               // alpha - 1 (no cancellation)

    // ---- Stage 1a: stage window [t0base-200, t0base+511] into LDS as f16 (float4 loads)
    for (int idx = tid; idx < B_ * (WCOLS / 4); idx += BLOCK) {
        const int row = idx / (WCOLS / 4);           // batch
        const int c   = (idx - row * (WCOLS / 4)) * 4;
        const int u   = t0base - HALO + c;           // global time (4-aligned; <0 region is 4-aligned too)
        v4f v = {};
        if (u >= 0)
            v = *(const v4f*)&x[((size_t)row * F_ + f) * S_ + u];
        v4h hv;
        hv[0] = (_Float16)v[0]; hv[1] = (_Float16)v[1];
        hv[2] = (_Float16)v[2]; hv[3] = (_Float16)v[3];
        *(v4h*)&xs[row * PITCH + c] = hv;
    }
    // ---- Stage 1b: zero the 16-column K-padding tail of each row
    for (int idx = tid; idx < B_ * (PITCH - WCOLS); idx += BLOCK) {
        const int row = idx >> 4;
        xs[row * PITCH + WCOLS + (idx & 15)] = (_Float16)0.0f;
    }
    // ---- Stage 1c: reciprocal running-weight table (depends on t only, not batch/lane)
    for (int c = tid; c < CHUNK; c += BLOCK) {
        const int t = t0base + c;
        const float n_eff = (float)((t < HALO ? t : HALO) + 1);
        const float tw = expm1f(n_eff * ln_a) / em1a;   // sum_{e=0}^{min(t,200)} alpha^e
        rden[c] = 1.0f / (tw + 1e-8f);
    }
    // ---- Stage 1d: Toeplitz A fragments (feature-constant), built cooperatively ONCE per block,
    //      stored directly as per-lane A 16x32 operands in this code's contiguous k order:
    //      lane L holds M = L%16; half h -> K = h + 16*(L/16)
    for (int j = tid; j < 32 * (KTILES * 16); j += BLOCK) {
        const int L   = j / (KTILES * 16);
        const int rem = j - L * (KTILES * 16);
        const int kt  = rem >> 4;
        const int h   = rem & 15;
        const int k   = kt * 32 + h + ((L >> 4) << 4);
        const int e   = (L & 15) + HALO - k;            // exponent; band mask
        const float av = (e >= 0 && e <= HALO) ? FAST_EXP2((float)e * log2a) : 0.0f;
        afrag_lds[L * APITCH + rem] = (_Float16)av;
    }
    __syncthreads();

    // ---- Stage 2: each wave picks up its 7 A fragments (14 ds_load_b128, held in VGPRs)
    const int i_m = lane & 15;
    const int grp = lane >> 4;
    v16h a_frag[KTILES];
    #pragma unroll
    for (int kt = 0; kt < KTILES; ++kt)
        a_frag[kt] = *(const v16h*)&afrag_lds[lane * APITCH + kt * 16];

    // ---- Stage 3: 64 tiles, 8 per wave; uniform compile-time trip count (EXEC stays all-ones)
    #pragma unroll 2
    for (int it = 0; it < TILES_PER_WAVE; ++it) {
        const int tile = wave + it * NWAVES;
        const int mt = tile >> 1;                     // time tile
        const int nt = tile & 1;                      // batch half
        const int c0 = mt * 16;                       // window base column

        v8f acc = {};
        // B 32x16 operand in this code's contiguous k order (same as A): lane L holds N = L%16; half h -> K = h + 16*(L/16)
        const int brow = nt * 16 + i_m;
        const _Float16* bbase = &xs[brow * PITCH + c0 + (grp << 4)];
        #pragma unroll
        for (int kt = 0; kt < KTILES; ++kt) {
            const v16h b_frag = *(const v16h*)(bbase + kt * 32);
            acc = __builtin_amdgcn_wmma_f32_16x16x32_f16(
                /*neg_a=*/false, a_frag[kt],
                /*neg_b=*/false, b_frag,
                /*c_mod=*/(short)0, acc,
                /*reuse_a=*/false, /*reuse_b=*/false);
        }

        // ---- Epilogue: multiply by precomputed reciprocal weights (broadcast LDS reads)
        // C/D layout: VGPR r holds (M = r + 8*(L/16), N = L%16)
        const int c_lo = mt * 16 + (grp << 3);        // 8 consecutive local timesteps
        const v4f rd0 = *(const v4f*)&rden[c_lo];
        const v4f rd1 = *(const v4f*)&rden[c_lo + 4];

        const int t_lo = t0base + c_lo;
        const int b    = nt * 16 + i_m;
        v4f lo, hi;
        lo[0] = acc[0] * rd0[0]; lo[1] = acc[1] * rd0[1];
        lo[2] = acc[2] * rd0[2]; lo[3] = acc[3] * rd0[3];
        hi[0] = acc[4] * rd1[0]; hi[1] = acc[5] * rd1[1];
        hi[2] = acc[6] * rd1[2]; hi[3] = acc[7] * rd1[3];

        float* op = out + ((size_t)b * F_ + f) * S_ + t_lo;
        *(v4f*)op       = lo;
        *((v4f*)op + 1) = hi;
    }
}

extern "C" void kernel_launch(void* const* d_in, const int* in_sizes, int n_in,
                              void* d_out, int out_size, void* d_ws, size_t ws_size,
                              hipStream_t stream) {
    const float* x   = (const float*)d_in[0];          // (B, F, S) f32
    const float* lhl = (const float*)d_in[1];          // (F,) f32
    float* out = (float*)d_out;                        // (B, F, S) f32
    (void)in_sizes; (void)n_in; (void)out_size; (void)d_ws; (void)ws_size;

    dim3 grid(F_ * NCHUNK);   // one workgroup per (feature, 512-step chunk)
    dim3 block(BLOCK);
    AdaptiveEMA_wmma_kernel<<<grid, block, 0, stream>>>(x, lhl, out);
}
